// PointDSCRework_44186623541797
// MI455X (gfx1250) — matrix operand order repaired and checked by reading, not compiled, not run on hardware
//
#include <hip/hip_runtime.h>
#include <hip/hip_bf16.h>
#include <math.h>

typedef __attribute__((ext_vector_type(16))) __bf16 v16bf;
typedef __attribute__((ext_vector_type(8)))  __bf16 v8bf;
typedef __attribute__((ext_vector_type(8)))  float  v8f;

#define F_BIAS  1
#define F_BN    2
#define F_RELU  4
#define F_RESID 8
#define F_OUTF  16
#define F_OUTB  32
#define F_OUTBT 64

#define SIGSP2_INV 100.0f   /* 1/(0.1^2) */

// ---------------- fragment loaders (CDNA5 WMMA bf16 layouts) ----------------
// A 16x32 bf16: lane half sel=(lane>>4): comps 0..7 -> K = k0+sel*8 .. +7,
//               comps 8..15 -> K = k0+sel*8+16 .. +23
__device__ inline v16bf load_fragA(const __bf16* row, int k0, int sel) {
  const v8bf lo = *reinterpret_cast<const v8bf*>(row + k0 + sel * 8);
  const v8bf hi = *reinterpret_cast<const v8bf*>(row + k0 + sel * 8 + 16);
  return __builtin_shufflevector(lo, hi, 0,1,2,3,4,5,6,7,8,9,10,11,12,13,14,15);
}
// B 32x16 bf16: lane holds column (lane&15), comps 0..7 -> K = k0+sel*8 .. +7,
//               comps 8..15 -> K = k0+sel*8+16 .. +23
__device__ inline v16bf load_fragB(const __bf16* row, int k0, int sel) {
  const v8bf lo = *reinterpret_cast<const v8bf*>(row + k0 + sel * 8);
  const v8bf hi = *reinterpret_cast<const v8bf*>(row + k0 + sel * 8 + 16);
  return __builtin_shufflevector(lo, hi, 0,1,2,3,4,5,6,7,8,9,10,11,12,13,14,15);
}

// -------- WMMA GEMM: Y[M,Nc] = epi( A[M,K] * Bw[Nc,K]^T ), 32x64 tile/wave --------
__global__ __launch_bounds__(128)
void k_gemm(const __bf16* __restrict__ A, const __bf16* __restrict__ Bw,
            const float* __restrict__ bias, const float* __restrict__ gam,
            const float* __restrict__ bet, const float* __restrict__ resid,
            float* __restrict__ outF, __bf16* __restrict__ outB,
            __bf16* __restrict__ outBT, int M, int Nc, int K, int flags)
{
  const int lane  = threadIdx.x & 31;
  const int wave  = threadIdx.x >> 5;
  const int tilesN = Nc >> 6;                // 64-wide tiles
  const int total  = (M >> 5) * tilesN;      // 32-tall tiles
  const int tile = blockIdx.x * 4 + wave;
  if (tile >= total) return;                 // wave-uniform: EXEC all-ones below
  const int tm = (tile / tilesN) << 5;
  const int tn = (tile % tilesN) << 6;
  const int r16 = lane & 15;
  const int sel = lane >> 4;

  const __bf16* a0p = A  + (size_t)(tm      + r16) * K;
  const __bf16* a1p = A  + (size_t)(tm + 16 + r16) * K;
  const __bf16* bp0 = Bw + (size_t)(tn      + r16) * K;
  const __bf16* bp1 = Bw + (size_t)(tn + 16 + r16) * K;
  const __bf16* bp2 = Bw + (size_t)(tn + 32 + r16) * K;
  const __bf16* bp3 = Bw + (size_t)(tn + 48 + r16) * K;

  v8f acc[2][4];
  #pragma unroll
  for (int i = 0; i < 2; i++)
    #pragma unroll
    for (int j = 0; j < 4; j++) acc[i][j] = (v8f){};

  for (int k0 = 0; k0 < K; k0 += 32) {
    v16bf a0 = load_fragA(a0p, k0, sel);
    v16bf a1 = load_fragA(a1p, k0, sel);
    v16bf b0 = load_fragB(bp0, k0, sel);
    v16bf b1 = load_fragB(bp1, k0, sel);
    v16bf b2 = load_fragB(bp2, k0, sel);
    v16bf b3 = load_fragB(bp3, k0, sel);
    if (k0 + 32 < K) {                       // gfx1250 prefetch, WGP scope (all levels)
      __builtin_prefetch(a0p + k0 + 32, 0, 3);
      __builtin_prefetch(a1p + k0 + 32, 0, 3);
      __builtin_prefetch(bp0 + k0 + 32, 0, 3);
      __builtin_prefetch(bp1 + k0 + 32, 0, 3);
      __builtin_prefetch(bp2 + k0 + 32, 0, 3);
      __builtin_prefetch(bp3 + k0 + 32, 0, 3);
    }
    acc[0][0] = __builtin_amdgcn_wmma_f32_16x16x32_bf16(false, a0, false, b0, (short)0, acc[0][0], false, false);
    acc[0][1] = __builtin_amdgcn_wmma_f32_16x16x32_bf16(false, a0, false, b1, (short)0, acc[0][1], false, false);
    acc[0][2] = __builtin_amdgcn_wmma_f32_16x16x32_bf16(false, a0, false, b2, (short)0, acc[0][2], false, false);
    acc[0][3] = __builtin_amdgcn_wmma_f32_16x16x32_bf16(false, a0, false, b3, (short)0, acc[0][3], false, false);
    acc[1][0] = __builtin_amdgcn_wmma_f32_16x16x32_bf16(false, a1, false, b0, (short)0, acc[1][0], false, false);
    acc[1][1] = __builtin_amdgcn_wmma_f32_16x16x32_bf16(false, a1, false, b1, (short)0, acc[1][1], false, false);
    acc[1][2] = __builtin_amdgcn_wmma_f32_16x16x32_bf16(false, a1, false, b2, (short)0, acc[1][2], false, false);
    acc[1][3] = __builtin_amdgcn_wmma_f32_16x16x32_bf16(false, a1, false, b3, (short)0, acc[1][3], false, false);
  }

  // Fused epilogue. D layout: vgpr r -> row r + 8*sel, col lane&15.
  const int rowoff = sel << 3;
  const float bnscale = rsqrtf(1.0f + 1e-5f);
  #pragma unroll
  for (int i = 0; i < 2; i++) {
    #pragma unroll
    for (int j = 0; j < 4; j++) {
      const int gc = tn + j * 16 + r16;
      float bi = (flags & F_BIAS) ? bias[gc] : 0.f;
      float sc = 1.f, sh = 0.f;
      if (flags & F_BN) { sc = gam[gc] * bnscale; sh = bet[gc]; }
      #pragma unroll
      for (int r = 0; r < 8; r++) {
        const int gr = tm + i * 16 + rowoff + r;
        float v = acc[i][j][r] + bi;
        if (flags & F_BN)   v = v * sc + sh;
        if (flags & F_RELU) v = fmaxf(v, 0.f);
        const size_t oi = (size_t)gr * Nc + gc;
        if (flags & F_RESID) v += resid[oi];
        if (flags & F_OUTF)  outF[oi] = v;
        if (flags & F_OUTB)  outB[oi] = (__bf16)v;
        if (flags & F_OUTBT) outBT[(size_t)gc * M + gr] = (__bf16)v;
      }
    }
  }
}

// ---------------- misc kernels ----------------
__global__ void k_f2bf(const float* __restrict__ s, __bf16* __restrict__ d, int n) {
  int i = blockIdx.x * 256 + threadIdx.x;
  if (i < n) d[i] = (__bf16)s[i];
}

__global__ __launch_bounds__(256)
void k_beta(const float* __restrict__ src, const float* __restrict__ tgt,
            float* __restrict__ beta, int N_)
{
  const int row = blockIdx.x, b = row / N_, i = row % N_;
  const float* s = src + (size_t)b * N_ * 3;
  const float* g = tgt + (size_t)b * N_ * 3;
  const float six = s[i*3], siy = s[i*3+1], siz = s[i*3+2];
  const float gix = g[i*3], giy = g[i*3+1], giz = g[i*3+2];
  float* out = beta + (size_t)row * N_;
  for (int j = threadIdx.x; j < N_; j += 256) {
    float dx = six - s[j*3], dy = siy - s[j*3+1], dz = siz - s[j*3+2];
    float ds = sqrtf(dx*dx + dy*dy + dz*dz);
    dx = gix - g[j*3]; dy = giy - g[j*3+1]; dz = giz - g[j*3+2];
    float dt = sqrtf(dx*dx + dy*dy + dz*dz);
    float ld = ds - dt;
    out[j] = fmaxf(1.f - ld * ld * SIGSP2_INV, 0.f);
  }
}

__global__ __launch_bounds__(128)
void k_layer0(const float* __restrict__ cors, const float* __restrict__ W,
              const float* __restrict__ b, float* __restrict__ xf, __bf16* __restrict__ xb)
{
  const int p = blockIdx.x, o = threadIdx.x;
  const float* cp = cors + (size_t)p * 6;
  const float* wr = W + (size_t)o * 6;
  float a = b[o];
  #pragma unroll
  for (int c = 0; c < 6; c++) a += cp[c] * wr[c];
  xf[(size_t)p * 128 + o] = a;
  xb[(size_t)p * 128 + o] = (__bf16)a;
}

// softmax over keys m of scores[n,m]*beta[n,m]; writes transposed bf16 wT[m,n]
__global__ __launch_bounds__(256)
void k_softmax_beta(const float* __restrict__ scores, const float* __restrict__ beta,
                    __bf16* __restrict__ wT, int N_)
{
  __shared__ float vals[2048];
  __shared__ float red[256];
  const int n = blockIdx.x, t = threadIdx.x;
  const float* sr = scores + (size_t)n * N_;
  const float* br = beta   + (size_t)n * N_;
  float mx = -3.4e38f;
  for (int m = t; m < N_; m += 256) { float v = sr[m] * br[m]; vals[m] = v; mx = fmaxf(mx, v); }
  red[t] = mx; __syncthreads();
  for (int s = 128; s > 0; s >>= 1) { if (t < s) red[t] = fmaxf(red[t], red[t + s]); __syncthreads(); }
  mx = red[0]; __syncthreads();
  float sum = 0.f;
  for (int m = t; m < N_; m += 256) { float e = expf(vals[m] - mx); vals[m] = e; sum += e; }
  red[t] = sum; __syncthreads();
  for (int s = 128; s > 0; s >>= 1) { if (t < s) red[t] += red[t + s]; __syncthreads(); }
  const float inv = 1.f / red[0];
  for (int m = t; m < N_; m += 256) wT[(size_t)m * N_ + n] = (__bf16)(vals[m] * inv);
}

__global__ __launch_bounds__(128)
void k_rownorm(const float* __restrict__ x, float* __restrict__ f)
{
  __shared__ float red[128];
  const int p = blockIdx.x, t = threadIdx.x;
  const float v = x[(size_t)p * 128 + t];
  red[t] = v * v; __syncthreads();
  for (int s = 64; s > 0; s >>= 1) { if (t < s) red[t] += red[t + s]; __syncthreads(); }
  const float inv = 1.f / fmaxf(sqrtf(red[0]), 1e-12f);
  f[(size_t)p * 128 + t] = v * inv;
}

__global__ __launch_bounds__(32)
void k_conf(const float* __restrict__ x, const float* __restrict__ W1, const float* __restrict__ b1,
            const float* __restrict__ W2, const float* __restrict__ b2,
            const float* __restrict__ W3, const float* __restrict__ b3, float* __restrict__ conf)
{
  __shared__ float c1[32], c2[32];
  const int p = blockIdx.x, t = threadIdx.x;
  const float* xp = x + (size_t)p * 128;
  float a = b1[t];
  for (int c = 0; c < 128; c++) a += W1[t * 128 + c] * xp[c];
  c1[t] = fmaxf(a, 0.f); __syncthreads();
  a = b2[t];
  for (int c = 0; c < 32; c++) a += W2[t * 32 + c] * c1[c];
  c2[t] = fmaxf(a, 0.f); __syncthreads();
  if (t == 0) { float s = b3[0]; for (int c = 0; c < 32; c++) s += W3[c] * c2[c]; conf[p] = s; }
}

// top-S selection, sorted desc, ties -> smaller index (matches lax.top_k)
__global__ __launch_bounds__(256)
void k_topk(const float* __restrict__ conf, int* __restrict__ seed_idx, int N_, int S_)
{
  __shared__ float vals[2048];
  __shared__ float rv[256]; __shared__ int ri[256];
  const int b = blockIdx.x, t = threadIdx.x;
  const float* cp = conf + (size_t)b * N_;
  for (int i = t; i < N_; i += 256) vals[i] = cp[i];
  __syncthreads();
  for (int s = 0; s < S_; s++) {
    float bv = -3.4e38f; int bi = 0x7fffffff;
    for (int i = t; i < N_; i += 256) {
      float v = vals[i];
      if (v > bv || (v == bv && i < bi)) { bv = v; bi = i; }
    }
    rv[t] = bv; ri[t] = bi; __syncthreads();
    for (int st = 128; st > 0; st >>= 1) {
      if (t < st && (rv[t+st] > rv[t] || (rv[t+st] == rv[t] && ri[t+st] < ri[t]))) { rv[t] = rv[t+st]; ri[t] = ri[t+st]; }
      __syncthreads();
    }
    if (t == 0) { seed_idx[b * S_ + s] = ri[0]; vals[ri[0]] = -3.4e38f; }
    __syncthreads();
  }
}

__global__ __launch_bounds__(256)
void k_seeddot(const float* __restrict__ f, const int* __restrict__ seed_idx,
               float* __restrict__ dot, int N_, int S_)
{
  __shared__ float fs[128];
  const int bs = blockIdx.x, b = bs / S_, s = bs % S_, t = threadIdx.x;
  const int si = seed_idx[b * S_ + s];
  const float* fb = f + (size_t)b * N_ * 128;
  if (t < 128) fs[t] = fb[(size_t)si * 128 + t];
  __syncthreads();
  float* dp = dot + (size_t)bs * N_;
  for (int n = t; n < N_; n += 256) {
    const float* fn = fb + (size_t)n * 128;
    float a = 0.f;
    for (int c = 0; c < 128; c++) a += fs[c] * fn[c];
    dp[n] = a;
  }
}

// top-(k+1) by inner product desc (== smallest feature distance), drop self
__global__ __launch_bounds__(256)
void k_knnsel(const float* __restrict__ dot, int* __restrict__ knn, int N_, int K_)
{
  __shared__ float vals[2048];
  __shared__ float rv[256]; __shared__ int ri[256];
  const int bs = blockIdx.x, t = threadIdx.x;
  const float* dp = dot + (size_t)bs * N_;
  for (int i = t; i < N_; i += 256) vals[i] = dp[i];
  __syncthreads();
  for (int s = 0; s <= K_; s++) {
    float bv = -3.4e38f; int bi = 0x7fffffff;
    for (int i = t; i < N_; i += 256) {
      float v = vals[i];
      if (v > bv || (v == bv && i < bi)) { bv = v; bi = i; }
    }
    rv[t] = bv; ri[t] = bi; __syncthreads();
    for (int st = 128; st > 0; st >>= 1) {
      if (t < st && (rv[t+st] > rv[t] || (rv[t+st] == rv[t] && ri[t+st] < ri[t]))) { rv[t] = rv[t+st]; ri[t] = ri[t+st]; }
      __syncthreads();
    }
    if (t == 0) { if (s > 0) knn[bs * K_ + (s - 1)] = ri[0]; vals[ri[0]] = -3.4e38f; }
    __syncthreads();
  }
}

__global__ __launch_bounds__(128)
void k_gather(const float* __restrict__ f, const float* __restrict__ src, const float* __restrict__ tgt,
              const int* __restrict__ knn, float* __restrict__ fk, float* __restrict__ sk,
              float* __restrict__ tk, int N_, int S_, int K_)
{
  const int g = blockIdx.x, t = threadIdx.x;
  const int b = g / (S_ * K_);
  const int idx = knn[g];
  const float* fb = f + (size_t)b * N_ * 128;
  fk[(size_t)g * 128 + t] = fb[(size_t)idx * 128 + t];
  if (t < 3) {
    sk[g * 3 + t] = src[((size_t)b * N_ + idx) * 3 + t];
    tk[g * 3 + t] = tgt[((size_t)b * N_ + idx) * 3 + t];
  }
}

__global__ __launch_bounds__(256)
void k_compat(const float* __restrict__ fk, const float* __restrict__ sk, const float* __restrict__ tk,
              const float* __restrict__ sigma, float* __restrict__ Mm, int K_)
{
  __shared__ float fs[40 * 128];
  __shared__ float ss[120], ts[120];
  const int bs = blockIdx.x, t = threadIdx.x;
  const float* fkp = fk + (size_t)bs * K_ * 128;
  for (int i = t; i < K_ * 128; i += 256) fs[i] = fkp[i];
  for (int i = t; i < K_ * 3; i += 256) { ss[i] = sk[(size_t)bs * K_ * 3 + i]; ts[i] = tk[(size_t)bs * K_ * 3 + i]; }
  __syncthreads();
  const float sg = sigma[0];
  const float inv_sg2 = 1.f / (sg * sg);
  float* Mp = Mm + (size_t)bs * K_ * K_;
  for (int p = t; p < K_ * K_; p += 256) {
    const int k = p / K_, l = p % K_;
    const float* f1 = fs + k * 128; const float* f2 = fs + l * 128;
    float a = 0.f;
    for (int c = 0; c < 128; c++) a += f1[c] * f2[c];
    float fm = fmaxf(1.f - (1.f - a) * inv_sg2, 0.f);
    float dx = ss[k*3] - ss[l*3], dy = ss[k*3+1] - ss[l*3+1], dz = ss[k*3+2] - ss[l*3+2];
    float dsn = sqrtf(dx*dx + dy*dy + dz*dz);
    dx = ts[k*3] - ts[l*3]; dy = ts[k*3+1] - ts[l*3+1]; dz = ts[k*3+2] - ts[l*3+2];
    float dtn = sqrtf(dx*dx + dy*dy + dz*dz);
    float d = dsn - dtn;
    float sm = fmaxf(1.f - d * d * SIGSP2_INV, 0.f);
    Mp[p] = (k == l) ? 0.f : fm * sm;
  }
}

__device__ inline float det3(const float M[3][3]) {
  return M[0][0]*(M[1][1]*M[2][2]-M[1][2]*M[2][1])
       - M[0][1]*(M[1][0]*M[2][2]-M[1][2]*M[2][0])
       + M[0][2]*(M[1][0]*M[2][1]-M[1][1]*M[2][0]);
}

__device__ void eig3(float A[3][3], float V[3][3]) {
  for (int i = 0; i < 3; i++) for (int j = 0; j < 3; j++) V[i][j] = (i == j) ? 1.f : 0.f;
  const int PP[3] = {0, 0, 1}, QQ[3] = {1, 2, 2};
  for (int sweep = 0; sweep < 20; sweep++) {
    for (int r = 0; r < 3; r++) {
      const int p = PP[r], q = QQ[r];
      float apq = A[p][q];
      if (fabsf(apq) < 1e-20f) continue;
      float theta = (A[q][q] - A[p][p]) / (2.f * apq);
      float tt = 1.f / (fabsf(theta) + sqrtf(theta * theta + 1.f));
      if (theta < 0.f) tt = -tt;
      float c = 1.f / sqrtf(tt * tt + 1.f), s = tt * c;
      for (int k = 0; k < 3; k++) { float a0 = A[k][p], a1 = A[k][q]; A[k][p] = c*a0 - s*a1; A[k][q] = s*a0 + c*a1; }
      for (int k = 0; k < 3; k++) { float a0 = A[p][k], a1 = A[q][k]; A[p][k] = c*a0 - s*a1; A[q][k] = s*a0 + c*a1; }
      for (int k = 0; k < 3; k++) { float v0 = V[k][p], v1 = V[k][q]; V[k][p] = c*v0 - s*v1; V[k][q] = s*v0 + c*v1; }
    }
  }
}

__global__ __launch_bounds__(64)
void k_power_kabsch(const float* __restrict__ Mm, const float* __restrict__ sk,
                    const float* __restrict__ tk, float* __restrict__ out, int K_)
{
  __shared__ float Ms[1600];
  __shared__ float vv[40], yy[40], ws[40];
  __shared__ float red0;
  const int bs = blockIdx.x, t = threadIdx.x;
  const float* Mp = Mm + (size_t)bs * K_ * K_;
  for (int i = t; i < K_ * K_; i += 64) Ms[i] = Mp[i];
  if (t < K_) vv[t] = 1.f;
  __syncthreads();
  for (int it = 0; it < 10; it++) {
    float y = 0.f;
    if (t < K_) { for (int l = 0; l < K_; l++) y += Ms[t * K_ + l] * vv[l]; yy[t] = y; }
    __syncthreads();
    if (t == 0) { float s = 0.f; for (int k = 0; k < K_; k++) s += yy[k] * yy[k]; red0 = sqrtf(s) + 1e-6f; }
    __syncthreads();
    if (t < K_) vv[t] = yy[t] / red0;
    __syncthreads();
  }
  if (t == 0) { float s = 0.f; for (int k = 0; k < K_; k++) s += vv[k]; red0 = s + 1e-6f; }
  __syncthreads();
  if (t < K_) ws[t] = fmaxf(vv[t] / red0, 0.f);
  __syncthreads();
  if (t != 0) return;

  const float* sp = sk + (size_t)bs * K_ * 3;
  const float* tp = tk + (size_t)bs * K_ * 3;
  float wsum = 0.f; for (int k = 0; k < K_; k++) wsum += ws[k];
  float cA[3] = {0,0,0}, cB[3] = {0,0,0};
  for (int k = 0; k < K_; k++) for (int d = 0; d < 3; d++) { cA[d] += sp[k*3+d]*ws[k]; cB[d] += tp[k*3+d]*ws[k]; }
  const float inv = 1.f / (wsum + 1e-6f);
  for (int d = 0; d < 3; d++) { cA[d] *= inv; cB[d] *= inv; }
  float Hm[3][3] = {{0,0,0},{0,0,0},{0,0,0}};
  for (int k = 0; k < K_; k++) {
    const float w = ws[k];
    const float a[3] = { sp[k*3]-cA[0], sp[k*3+1]-cA[1], sp[k*3+2]-cA[2] };
    const float b[3] = { tp[k*3]-cB[0], tp[k*3+1]-cB[1], tp[k*3+2]-cB[2] };
    for (int i = 0; i < 3; i++) for (int j = 0; j < 3; j++) Hm[i][j] += a[i] * w * b[j];
  }
  // SVD of H via Jacobi eig of H^T H
  float A[3][3], V[3][3];
  for (int i = 0; i < 3; i++) for (int j = 0; j < 3; j++) {
    float s = 0.f; for (int k = 0; k < 3; k++) s += Hm[k][i] * Hm[k][j]; A[i][j] = s;
  }
  eig3(A, V);
  float lam[3] = { A[0][0], A[1][1], A[2][2] };
  for (int i = 0; i < 2; i++) for (int j = i + 1; j < 3; j++) if (lam[j] > lam[i]) {
    float tl = lam[i]; lam[i] = lam[j]; lam[j] = tl;
    for (int k = 0; k < 3; k++) { float tv = V[k][i]; V[k][i] = V[k][j]; V[k][j] = tv; }
  }
  float u0[3], u1[3], u2[3];
  for (int i = 0; i < 3; i++) { u0[i] = Hm[i][0]*V[0][0] + Hm[i][1]*V[1][0] + Hm[i][2]*V[2][0]; }
  float n0 = sqrtf(u0[0]*u0[0] + u0[1]*u0[1] + u0[2]*u0[2]);
  if (n0 > 1e-12f) { for (int i = 0; i < 3; i++) u0[i] /= n0; } else { u0[0]=1.f; u0[1]=0.f; u0[2]=0.f; }
  for (int i = 0; i < 3; i++) { u1[i] = Hm[i][0]*V[0][1] + Hm[i][1]*V[1][1] + Hm[i][2]*V[2][1]; }
  float d01 = u0[0]*u1[0] + u0[1]*u1[1] + u0[2]*u1[2];
  for (int i = 0; i < 3; i++) u1[i] -= d01 * u0[i];
  float n1 = sqrtf(u1[0]*u1[0] + u1[1]*u1[1] + u1[2]*u1[2]);
  if (n1 > 1e-12f) { for (int i = 0; i < 3; i++) u1[i] /= n1; }
  else {
    float ax[3] = { 1.f, 0.f, 0.f };
    if (fabsf(u0[0]) > 0.9f) { ax[0] = 0.f; ax[1] = 1.f; }
    u1[0] = u0[1]*ax[2] - u0[2]*ax[1]; u1[1] = u0[2]*ax[0] - u0[0]*ax[2]; u1[2] = u0[0]*ax[1] - u0[1]*ax[0];
    float nn = sqrtf(u1[0]*u1[0] + u1[1]*u1[1] + u1[2]*u1[2]) + 1e-20f;
    for (int i = 0; i < 3; i++) u1[i] /= nn;
  }
  u2[0] = u0[1]*u1[2] - u0[2]*u1[1];
  u2[1] = u0[2]*u1[0] - u0[0]*u1[2];
  u2[2] = u0[0]*u1[1] - u0[1]*u1[0];          // det(U)=+1 by construction
  const float delta = det3(V);                 // delta = det(V U^T) = det(V)
  float R[3][3];
  for (int i = 0; i < 3; i++) for (int k = 0; k < 3; k++)
    R[i][k] = V[i][0]*u0[k] + V[i][1]*u1[k] + delta * V[i][2]*u2[k];
  float tr[3];
  for (int i = 0; i < 3; i++) tr[i] = cB[i] - (R[i][0]*cA[0] + R[i][1]*cA[1] + R[i][2]*cA[2]);
  float* op = out + (size_t)bs * 16;
  for (int i = 0; i < 3; i++) { for (int j = 0; j < 3; j++) op[i*4+j] = R[i][j]; op[i*4+3] = tr[i]; }
  op[12] = 0.f; op[13] = 0.f; op[14] = 0.f; op[15] = 1.f;
}

// ---------------- host side ----------------
static inline void launch_gemm(hipStream_t st, const __bf16* A, const __bf16* Bw,
                               const float* bias, const float* g, const float* be,
                               const float* resid, float* oF, __bf16* oB, __bf16* oT,
                               int M, int Nc, int K, int flags)
{
  const int tiles = (M / 32) * (Nc / 64);
  const int blocks = (tiles + 3) / 4;
  k_gemm<<<blocks, 128, 0, st>>>(A, Bw, bias, g, be, resid, oF, oB, oT, M, Nc, K, flags);
}

extern "C" void kernel_launch(void* const* d_in, const int* in_sizes, int n_in,
                              void* d_out, int out_size, void* d_ws, size_t ws_size,
                              hipStream_t stream)
{
  (void)in_sizes; (void)n_in; (void)out_size; (void)ws_size;
  const int B = 2, N = 2048, C = 128, Hh = 64, BN = 4096, S = 204, KN = 40;

  const float* cors  = (const float*)d_in[0];
  const float* src   = (const float*)d_in[1];
  const float* tgt   = (const float*)d_in[2];
  const float* W0    = (const float*)d_in[3];
  const float* b0    = (const float*)d_in[4];
  const float* sigma = (const float*)d_in[5];
  auto P = [&](int i) { return (const float*)d_in[i]; };

  char* ws = (char*)d_ws; size_t cur = 0;
  auto alloc = [&](size_t bytes) -> void* {
    cur = (cur + 255) & ~(size_t)255; void* p = ws + cur; cur += bytes; return p;
  };

  float*  xf    = (float*) alloc((size_t)BN * C * 4);
  __bf16* xbA   = (__bf16*)alloc((size_t)BN * C * 2);
  __bf16* xbB   = (__bf16*)alloc((size_t)BN * C * 2);
  __bf16* pcnb  = (__bf16*)alloc((size_t)BN * C * 2);
  __bf16* qbf   = (__bf16*)alloc((size_t)BN * C * 2);
  __bf16* kbf   = (__bf16*)alloc((size_t)BN * C * 2);
  __bf16* vT    = (__bf16*)alloc((size_t)N * C * 2);       // per-batch, reused
  float*  scores= (float*) alloc((size_t)N * N * 4);       // per-batch, reused
  __bf16* wTb   = (__bf16*)alloc((size_t)N * N * 2);       // per-batch, reused
  __bf16* msgb  = (__bf16*)alloc((size_t)BN * C * 2);
  __bf16* h1b   = (__bf16*)alloc((size_t)BN * Hh * 2);
  __bf16* h2b   = (__bf16*)alloc((size_t)BN * Hh * 2);
  float*  beta  = (float*) alloc((size_t)B * N * N * 4);
  float*  ff    = (float*) alloc((size_t)BN * C * 4);
  float*  conf  = (float*) alloc((size_t)BN * 4);
  int*    seedi = (int*)   alloc((size_t)B * S * 4);
  float*  dotb  = (float*) alloc((size_t)B * S * N * 4);
  int*    knn   = (int*)   alloc((size_t)B * S * KN * 4);
  float*  fk    = (float*) alloc((size_t)B * S * KN * C * 4);
  float*  skn   = (float*) alloc((size_t)B * S * KN * 3 * 4);
  float*  tkn   = (float*) alloc((size_t)B * S * KN * 3 * 4);
  float*  Mm    = (float*) alloc((size_t)B * S * KN * KN * 4);

  __bf16 *wpcn[6], *wq[6], *wk[6], *wv[6], *w1[6], *w2[6], *w3[6];
  for (int i = 0; i < 6; i++) {
    wpcn[i] = (__bf16*)alloc((size_t)C * C * 2);
    wq[i]   = (__bf16*)alloc((size_t)C * C * 2);
    wk[i]   = (__bf16*)alloc((size_t)C * C * 2);
    wv[i]   = (__bf16*)alloc((size_t)C * C * 2);
    w1[i]   = (__bf16*)alloc((size_t)Hh * C * 2);
    w2[i]   = (__bf16*)alloc((size_t)Hh * Hh * 2);
    w3[i]   = (__bf16*)alloc((size_t)C * Hh * 2);
  }
  auto cvt = [&](const float* s, __bf16* d, int n) {
    k_f2bf<<<(n + 255) / 256, 256, 0, stream>>>(s, d, n);
  };
  for (int i = 0; i < 6; i++) {
    const int base = 6 + i * 20;
    cvt(P(base + 0),  wpcn[i], C * C);
    cvt(P(base + 4),  wq[i],   C * C);
    cvt(P(base + 6),  wk[i],   C * C);
    cvt(P(base + 8),  wv[i],   C * C);
    cvt(P(base + 10), w1[i],   Hh * C);
    cvt(P(base + 14), w2[i],   Hh * Hh);
    cvt(P(base + 18), w3[i],   C * Hh);
  }

  k_beta<<<B * N, 256, 0, stream>>>(src, tgt, beta, N);
  k_layer0<<<BN, 128, 0, stream>>>(cors, W0, b0, xf, xbA);

  __bf16* xcur = xbA; __bf16* xnxt = xbB;
  for (int i = 0; i < 6; i++) {
    const int base = 6 + i * 20;
    // x = relu(bn(pcn(x)))  -> xf (f32 master) + pcnb (bf16)
    launch_gemm(stream, xcur, wpcn[i], P(base + 1), P(base + 2), P(base + 3), nullptr,
                xf, pcnb, nullptr, BN, C, C, F_BIAS | F_BN | F_RELU | F_OUTF | F_OUTB);
    // Q, K
    launch_gemm(stream, pcnb, wq[i], P(base + 5), nullptr, nullptr, nullptr,
                nullptr, qbf, nullptr, BN, C, C, F_BIAS | F_OUTB);
    launch_gemm(stream, pcnb, wk[i], P(base + 7), nullptr, nullptr, nullptr,
                nullptr, kbf, nullptr, BN, C, C, F_BIAS | F_OUTB);
    for (int b = 0; b < B; b++) {
      const size_t po = (size_t)b * N * C;
      // V (bf16, stored transposed [C,N] for the message GEMM)
      launch_gemm(stream, pcnb + po, wv[i], P(base + 9), nullptr, nullptr, nullptr,
                  nullptr, nullptr, vT, N, C, C, F_BIAS | F_OUTBT);
      // scores = Q K^T  (f32)
      launch_gemm(stream, qbf + po, kbf + po, nullptr, nullptr, nullptr, nullptr,
                  scores, nullptr, nullptr, N, N, C, F_OUTF);
      // w = softmax(scores * beta), stored transposed bf16
      k_softmax_beta<<<N, 256, 0, stream>>>(scores, beta + (size_t)b * N * N, wTb, N);
      // msg = w^T V
      launch_gemm(stream, wTb, vT, nullptr, nullptr, nullptr, nullptr,
                  nullptr, msgb + po, nullptr, N, C, N, F_OUTB);
    }
    // MLP + residual
    launch_gemm(stream, msgb, w1[i], P(base + 11), P(base + 12), P(base + 13), nullptr,
                nullptr, h1b, nullptr, BN, Hh, C, F_BIAS | F_BN | F_RELU | F_OUTB);
    launch_gemm(stream, h1b, w2[i], P(base + 15), P(base + 16), P(base + 17), nullptr,
                nullptr, h2b, nullptr, BN, Hh, Hh, F_BIAS | F_BN | F_RELU | F_OUTB);
    launch_gemm(stream, h2b, w3[i], P(base + 19), nullptr, nullptr, xf,
                xf, xnxt, nullptr, BN, C, Hh, F_BIAS | F_RESID | F_OUTF | F_OUTB);
    __bf16* tmp = xcur; xcur = xnxt; xnxt = tmp;
  }

  k_rownorm<<<BN, 128, 0, stream>>>(xf, ff);
  k_conf<<<BN, 32, 0, stream>>>(xf, P(126), P(127), P(128), P(129), P(130), P(131), conf);
  k_topk<<<B, 256, 0, stream>>>(conf, seedi, N, S);
  k_seeddot<<<B * S, 256, 0, stream>>>(ff, seedi, dotb, N, S);
  k_knnsel<<<B * S, 256, 0, stream>>>(dotb, knn, N, KN);
  k_gather<<<B * S * KN, 128, 0, stream>>>(ff, src, tgt, knn, fk, skn, tkn, N, S, KN);
  k_compat<<<B * S, 256, 0, stream>>>(fk, skn, tkn, sigma, Mm, KN);
  k_power_kabsch<<<B * S, 64, 0, stream>>>(Mm, skn, tkn, (float*)d_out, KN);
}
